// Seq2SeqWithAttention_48232482734889
// MI455X (gfx1250) — matrix operand order repaired and checked by reading, not compiled, not run on hardware
//
#include <hip/hip_runtime.h>
#include <hip/hip_fp16.h>

// ---------------------------------------------------------------------------
// Seq2Seq w/ attention (GRU enc/dec) for gfx1250 (MI455X), wave32 + WMMA.
//   S=512, B=32, E=H=256, V=32000, T=64.  All GEMMs: f16 in, f32 accumulate
//   via v_wmma_f32_16x16x32_f16.  Recurrences run as persistent single-WGP
//   kernels with h + gates staged in LDS (~144KB of the 320KB/WGP).
// ---------------------------------------------------------------------------

typedef __attribute__((ext_vector_type(16))) _Float16 v16h;
typedef __attribute__((ext_vector_type(8)))  float    v8f;

#define S_LEN 512
#define BATCH 32
#define EMB   256
#define HID   256
#define VOCAB 32000
#define TOUT  64

// ---- WMMA 16x16 tile: C += A[M,K] @ W[N,K]^T, f16 inputs, f32 accum -------
// Layouts per CDNA5 ISA 7.12.2 (wave32):
//   A 16x32 f16: lane l holds row m=l&15; K elems {k..k+7, k+16..k+23} for
//                l<16, {k+8..k+15, k+24..k+31} for l>=16.
//   B 32x16 f16: lane l holds col n=l&15; same K order as A.
//   C/D f32:     vgpr r -> m = r + 8*(l>>4), n = l&15.
template <int KT>
__device__ __forceinline__ v8f wmma_tile_f16(const _Float16* __restrict__ A, int lda,
                                             const _Float16* __restrict__ W, int ldw,
                                             int m0, int n0) {
  const int lane = threadIdx.x & 31;
  const int half = lane >> 4;
  const int l15  = lane & 15;
  const _Float16* __restrict__ arow = A + (size_t)(m0 + l15) * lda + half * 8;
  const _Float16* __restrict__ wrow = W + (size_t)(n0 + l15) * ldw + half * 8;
  v8f acc = {};
#pragma unroll
  for (int k = 0; k < KT; k += 32) {
    v16h a, b;
    const _Float16* ap = arow + k;
    const _Float16* bp = wrow + k;
    __builtin_prefetch(bp + 64, 0, 1);   // global_prefetch_b8 (next K chunk)
#pragma unroll
    for (int e = 0; e < 8; ++e) { a[e] = ap[e]; a[e + 8] = ap[e + 16]; }
#pragma unroll
    for (int e = 0; e < 8; ++e) { b[e] = bp[e]; b[e + 8] = bp[e + 16]; }
    acc = __builtin_amdgcn_wmma_f32_16x16x32_f16(false, a, false, b,
                                                 (short)0, acc, false, false);
  }
  return acc;
}

__device__ __forceinline__ void store_tile(v8f acc, const float* __restrict__ bias,
                                           float* __restrict__ C, _Float16* __restrict__ C16,
                                           int ldc, int m0, int n0) {
  const int lane = threadIdx.x & 31;
  const int half = lane >> 4;
  const int n = n0 + (lane & 15);
  const float bv = bias ? bias[n] : 0.0f;
#pragma unroll
  for (int r = 0; r < 8; ++r) {
    const int m = m0 + r + 8 * half;
    const float v = acc[r] + bv;
    C[(size_t)m * ldc + n] = v;
    if (C16) C16[(size_t)m * ldc + n] = (_Float16)v;
  }
}

// ---- generic GEMM: C[M,N] = A[M,K] @ W[N,K]^T (+bias), 1 tile / wave ------
template <int KT>
__global__ void k_gemm(const _Float16* __restrict__ A, const _Float16* __restrict__ W,
                       const float* __restrict__ bias,
                       float* __restrict__ C, _Float16* __restrict__ C16,
                       int M, int N) {
  const int wave  = blockIdx.x * (blockDim.x >> 5) + (threadIdx.x >> 5);
  const int ntn   = N >> 4;
  const int total = (M >> 4) * ntn;
  if (wave >= total) return;                    // wave-uniform: EXEC stays full
  const int m0 = (wave / ntn) << 4;
  const int n0 = (wave % ntn) << 4;
  v8f acc = wmma_tile_f16<KT>(A, KT, W, KT, m0, n0);
  store_tile(acc, bias, C, C16, N, m0, n0);
}

// ---- persistent GRU recurrence: one 1024-thread WG, h + gates in LDS ------
// gi: [S,B,768] precomputed x@Wih^T+bih.  Per step: gh = h@Whh^T+bhh (WMMA,
// 96 tiles over 32 waves), then fused gate math.  y written at stride ostride.
__global__ void k_gru_seq(const float* __restrict__ gi,
                          const _Float16* __restrict__ whh,
                          const float* __restrict__ bhh,
                          _Float16* __restrict__ y16, float* __restrict__ y32,
                          int ostride,
                          _Float16* __restrict__ hfin16, int hstride, int reverse) {
  extern __shared__ char smem[];
  _Float16* h16   = (_Float16*)smem;                                   // 32x256 f16
  float*    hprev = (float*)(smem + BATCH * HID * 2);                  // 32x256 f32
  float*    gh    = (float*)(smem + BATCH * HID * 2 + BATCH * HID * 4);// 32x768 f32
  const int tid  = threadIdx.x;
  const int wave = tid >> 5;
  const int nw   = blockDim.x >> 5;

  for (int i = tid; i < BATCH * HID; i += blockDim.x) {
    h16[i] = (_Float16)0.0f; hprev[i] = 0.0f;
  }
  __syncthreads();

  for (int step = 0; step < S_LEN; ++step) {
    const int t = reverse ? (S_LEN - 1 - step) : step;
    const float* __restrict__ git = gi + (size_t)t * BATCH * (3 * HID);

    // gh = h @ whh^T + bhh   (2 x 48 = 96 tiles)
    for (int tt = wave; tt < 96; tt += nw) {
      const int m0 = (tt & 1) << 4;
      const int n0 = (tt >> 1) << 4;
      v8f acc = wmma_tile_f16<HID>(h16, HID, whh, HID, m0, n0);
      const int lane = tid & 31;
      const int half = lane >> 4;
      const int n = n0 + (lane & 15);
      const float bv = bhh[n];
#pragma unroll
      for (int r = 0; r < 8; ++r)
        gh[(m0 + r + 8 * half) * (3 * HID) + n] = acc[r] + bv;
    }
    __syncthreads();

    // fused GRU gates
    for (int idx = tid; idx < BATCH * HID; idx += blockDim.x) {
      const int b = idx >> 8, j = idx & 255;
      const float* g  = git + b * (3 * HID);
      const float* gb = gh  + b * (3 * HID);
      const float r  = 1.0f / (1.0f + __expf(-(g[j]       + gb[j])));
      const float z  = 1.0f / (1.0f + __expf(-(g[256 + j] + gb[256 + j])));
      const float nn = tanhf(g[512 + j] + r * gb[512 + j]);
      const float h  = (1.0f - z) * nn + z * hprev[idx];
      hprev[idx] = h;
      h16[idx]   = (_Float16)h;
      const size_t o = ((size_t)t * BATCH + b) * (size_t)ostride + j;
      y16[o] = (_Float16)h;
      if (y32) y32[o] = h;
    }
    __syncthreads();
  }

  for (int idx = tid; idx < BATCH * HID; idx += blockDim.x) {
    const int b = idx >> 8, j = idx & 255;
    hfin16[b * hstride + j] = (_Float16)hprev[idx];
  }
}

// ---- small helpers --------------------------------------------------------
__global__ void k_cast(const float* __restrict__ s, _Float16* __restrict__ d, int n) {
  const int i = blockIdx.x * blockDim.x + threadIdx.x;
  if (i < n) d[i] = (_Float16)s[i];
}
// dst[C,R] = src[R,C]^T  (weight_matrix -> W[N,K] form)
__global__ void k_cast_t(const float* __restrict__ s, _Float16* __restrict__ d, int R, int C) {
  const int i = blockIdx.x * blockDim.x + threadIdx.x;
  if (i < R * C) { const int r = i / C, c = i % C; d[c * R + r] = (_Float16)s[i]; }
}
__global__ void k_embed(const int* __restrict__ X, const float* __restrict__ tab,
                        _Float16* __restrict__ out, int rows, int E) {
  const int i = blockIdx.x * blockDim.x + threadIdx.x;
  if (i < rows * E) {
    const int r = i / E, e = i - r * E;
    out[i] = (_Float16)tab[(size_t)X[r] * E + e];
  }
}
__global__ void k_fill_i32(int* __restrict__ p, int v, int n) {
  const int i = blockIdx.x * blockDim.x + threadIdx.x;
  if (i < n) p[i] = v;
}
// scores[s,b] = sum_e tanh(att_enc[s,b,e] + att_dec[b,e]) * value[e]
__global__ void k_scores(const float* __restrict__ ae, const float* __restrict__ ad,
                         const float* __restrict__ val, float* __restrict__ sc) {
  const int i = blockIdx.x * blockDim.x + threadIdx.x;
  if (i >= S_LEN * BATCH) return;
  const int b = i & (BATCH - 1);
  const float* a = ae + (size_t)i * EMB;
  const float* d = ad + b * EMB;
  float s = 0.0f;
  for (int e = 0; e < EMB; ++e) s += tanhf(a[e] + d[e]) * val[e];
  sc[i] = s;
}
// softmax over seq dim (one block per batch column, 512 threads)
__global__ void k_softmax(const float* __restrict__ sc, float* __restrict__ attn) {
  __shared__ float red[S_LEN];
  const int b = blockIdx.x, s = threadIdx.x;
  const float v = sc[s * BATCH + b];
  red[s] = v; __syncthreads();
  for (int off = S_LEN / 2; off > 0; off >>= 1) {
    if (s < off) red[s] = fmaxf(red[s], red[s + off]);
    __syncthreads();
  }
  const float m = red[0]; __syncthreads();
  const float e = __expf(v - m);
  red[s] = e; __syncthreads();
  for (int off = S_LEN / 2; off > 0; off >>= 1) {
    if (s < off) red[s] += red[s + off];
    __syncthreads();
  }
  attn[s * BATCH + b] = e / red[0];
}
// Xa[b,d] = sum_s attn[s,b] * enc_out[s,b,d]   (block per b, 512 threads = d)
__global__ void k_xa(const float* __restrict__ attn, const float* __restrict__ enc,
                     float* __restrict__ Xa) {
  const int b = blockIdx.x, d = threadIdx.x;
  float s = 0.0f;
  for (int t = 0; t < S_LEN; ++t)
    s += attn[t * BATCH + b] * enc[((size_t)t * BATCH + b) * (2 * HID) + d];
  Xa[b * (2 * HID) + d] = s;
}
// xcat[b] = concat(dec_emb[tok[b]], Xa[b])  -> f16  (32 x 768)
__global__ void k_xcat(const int* __restrict__ tok, const float* __restrict__ demb,
                       const float* __restrict__ Xa, _Float16* __restrict__ x) {
  const int i = blockIdx.x * blockDim.x + threadIdx.x;
  if (i >= BATCH * (EMB + 2 * HID)) return;
  const int b = i / (EMB + 2 * HID), j = i - b * (EMB + 2 * HID);
  const float v = (j < EMB) ? demb[(size_t)tok[b] * EMB + j]
                            : Xa[b * (2 * HID) + (j - EMB)];
  x[i] = (_Float16)v;
}
// GRU gate math for decoder cells (h updated in place, f32 + f16 copies)
__global__ void k_gate(const float* __restrict__ gi, const float* __restrict__ gh,
                       float* __restrict__ h, _Float16* __restrict__ h16) {
  const int idx = blockIdx.x * blockDim.x + threadIdx.x;
  if (idx >= BATCH * HID) return;
  const int b = idx >> 8, j = idx & 255;
  const float* gib = gi + b * (3 * HID);
  const float* ghb = gh + b * (3 * HID);
  const float r  = 1.0f / (1.0f + __expf(-(gib[j]       + ghb[j])));
  const float z  = 1.0f / (1.0f + __expf(-(gib[256 + j] + ghb[256 + j])));
  const float nn = tanhf(gib[512 + j] + r * ghb[512 + j]);
  const float hv = (1.0f - z) * nn + z * h[idx];
  h[idx] = hv; h16[idx] = (_Float16)hv;
}
// greedy argmax over vocab (block per b), first-max tie-break like jnp.argmax
__global__ void k_argmax(const float* __restrict__ logits, int* __restrict__ tok) {
  __shared__ float bv[256];
  __shared__ int   bi[256];
  const int b = blockIdx.x, t = threadIdx.x;
  const float* row = logits + (size_t)b * VOCAB;
  float best = -3.4e38f; int bidx = 0;
  for (int i = t; i < VOCAB; i += 256) {
    const float v = row[i];
    if (v > best) { best = v; bidx = i; }
  }
  bv[t] = best; bi[t] = bidx; __syncthreads();
  for (int off = 128; off > 0; off >>= 1) {
    if (t < off) {
      if (bv[t + off] > bv[t] || (bv[t + off] == bv[t] && bi[t + off] < bi[t])) {
        bv[t] = bv[t + off]; bi[t] = bi[t + off];
      }
    }
    __syncthreads();
  }
  if (t == 0) tok[b] = bi[0];
}

// ---------------------------------------------------------------------------
extern "C" void kernel_launch(void* const* d_in, const int* in_sizes, int n_in,
                              void* d_out, int out_size, void* d_ws, size_t ws_size,
                              hipStream_t stream) {
  (void)in_sizes; (void)n_in; (void)out_size; (void)ws_size;

  const int*   X        = (const int*)  d_in[0];
  const float* enc_emb  = (const float*)d_in[1];
  const float* dec_emb  = (const float*)d_in[2];
  const float* enc_wih0 = (const float*)d_in[3];
  const float* enc_whh0 = (const float*)d_in[4];
  const float* enc_bih0 = (const float*)d_in[5];
  const float* enc_bhh0 = (const float*)d_in[6];
  const float* enc_wih1 = (const float*)d_in[7];
  const float* enc_whh1 = (const float*)d_in[8];
  const float* enc_bih1 = (const float*)d_in[9];
  const float* enc_bhh1 = (const float*)d_in[10];
  const float* enc_out_w= (const float*)d_in[11];
  const float* dec_wih  = (const float*)d_in[12];
  const float* dec_whh  = (const float*)d_in[13];
  const float* dec_bih  = (const float*)d_in[14];
  const float* dec_bhh  = (const float*)d_in[15];
  const float* dec_out_b= (const float*)d_in[16];
  const float* weight_m = (const float*)d_in[17];
  const float* value_v  = (const float*)d_in[18];
  const float* l2_w     = (const float*)d_in[19];
  const float* l2_b     = (const float*)d_in[20];
  const float* l3_w     = (const float*)d_in[21];
  const float* l3_b     = (const float*)d_in[22];
  float* out = (float*)d_out;

  // ---- carve workspace ----------------------------------------------------
  char* p = (char*)d_ws;
  auto take = [&](size_t bytes) { char* r = p; p += (bytes + 255) & ~(size_t)255; return r; };
  const int H3 = 3 * HID;   // 768
  const int SB = S_LEN * BATCH;

  _Float16* w_wih0 = (_Float16*)take((size_t)2 * H3 * EMB * 2);
  _Float16* w_whh0 = (_Float16*)take((size_t)2 * H3 * HID * 2);
  _Float16* w_wih1 = (_Float16*)take((size_t)2 * H3 * 2 * HID * 2);
  _Float16* w_whh1 = (_Float16*)take((size_t)2 * H3 * HID * 2);
  _Float16* w_eow  = (_Float16*)take((size_t)EMB * 2 * HID * 2);
  _Float16* w_wmt  = (_Float16*)take((size_t)EMB * 2 * HID * 2);  // weight_matrix^T
  _Float16* w_dwih = (_Float16*)take((size_t)2 * H3 * EMB * 2);
  _Float16* w_dwhh = (_Float16*)take((size_t)2 * H3 * EMB * 2);
  _Float16* w_l2   = (_Float16*)take((size_t)EMB * (2 * HID + EMB) * 2);
  _Float16* w_l3   = (_Float16*)take((size_t)EMB * EMB * 2);
  _Float16* w_demb = (_Float16*)take((size_t)VOCAB * EMB * 2);

  _Float16* emb16  = (_Float16*)take((size_t)SB * EMB * 2);
  _Float16* y0_16  = (_Float16*)take((size_t)SB * 2 * HID * 2);
  _Float16* enc16  = (_Float16*)take((size_t)SB * 2 * HID * 2);
  _Float16* hcat16 = (_Float16*)take((size_t)64 * 2 * HID * 2);
  _Float16* hid16  = (_Float16*)take((size_t)64 * EMB * 2);       // h0_16 | h1_16
  _Float16* xcat16 = (_Float16*)take((size_t)BATCH * (EMB + 2 * HID) * 2);
  _Float16* x16    = (_Float16*)take((size_t)BATCH * EMB * 2);

  float* gi0f   = (float*)take((size_t)SB * H3 * 4);
  float* gi0b   = (float*)take((size_t)SB * H3 * 4);
  float* gi1f   = (float*)take((size_t)SB * H3 * 4);
  float* gi1b   = (float*)take((size_t)SB * H3 * 4);
  float* enc32  = (float*)take((size_t)SB * 2 * HID * 4);
  float* attenc = (float*)take((size_t)SB * EMB * 4);
  float* hid32  = (float*)take((size_t)64 * EMB * 4);             // h0_32 | h1_32
  float* attdec = (float*)take((size_t)BATCH * EMB * 4);
  float* scores = (float*)take((size_t)SB * 4);
  float* attn   = (float*)take((size_t)SB * 4);
  float* Xa     = (float*)take((size_t)BATCH * 2 * HID * 4);
  float* x32    = (float*)take((size_t)BATCH * EMB * 4);
  float* gi_d   = (float*)take((size_t)BATCH * H3 * 4);
  float* gh_d   = (float*)take((size_t)BATCH * H3 * 4);
  int*   tok    = (int*)  take((size_t)BATCH * 4);

  float*    h0_32 = hid32;           float*    h1_32 = hid32 + BATCH * EMB;
  _Float16* h0_16 = hid16;           _Float16* h1_16 = hid16 + BATCH * EMB;

  auto cast = [&](const float* s, _Float16* d, int n) {
    k_cast<<<(n + 255) / 256, 256, 0, stream>>>(s, d, n);
  };
  auto gemm = [&](const _Float16* A, const _Float16* W, const float* bias,
                  float* C, _Float16* C16, int M, int N, int K) {
    const int blocks = ((M / 16) * (N / 16) + 7) / 8;
    if (K == 256)      k_gemm<256><<<blocks, 256, 0, stream>>>(A, W, bias, C, C16, M, N);
    else if (K == 512) k_gemm<512><<<blocks, 256, 0, stream>>>(A, W, bias, C, C16, M, N);
    else               k_gemm<768><<<blocks, 256, 0, stream>>>(A, W, bias, C, C16, M, N);
  };

  // ---- weight casts -------------------------------------------------------
  cast(enc_wih0, w_wih0, 2 * H3 * EMB);
  cast(enc_whh0, w_whh0, 2 * H3 * HID);
  cast(enc_wih1, w_wih1, 2 * H3 * 2 * HID);
  cast(enc_whh1, w_whh1, 2 * H3 * HID);
  cast(enc_out_w, w_eow, EMB * 2 * HID);
  k_cast_t<<<(2 * HID * EMB + 255) / 256, 256, 0, stream>>>(weight_m, w_wmt, 2 * HID, EMB);
  cast(dec_wih, w_dwih, 2 * H3 * EMB);
  cast(dec_whh, w_dwhh, 2 * H3 * EMB);
  cast(l2_w, w_l2, EMB * (2 * HID + EMB));
  cast(l3_w, w_l3, EMB * EMB);
  cast(dec_emb, w_demb, VOCAB * EMB);

  // ---- encoder ------------------------------------------------------------
  k_embed<<<(SB * EMB + 255) / 256, 256, 0, stream>>>(X, enc_emb, emb16, SB, EMB);

  // layer-0 input projections hoisted over all timesteps (M=16384)
  gemm(emb16, w_wih0,            enc_bih0,      gi0f, nullptr, SB, H3, EMB);
  gemm(emb16, w_wih0 + H3 * EMB, enc_bih0 + H3, gi0b, nullptr, SB, H3, EMB);

  const size_t gru_lds = (size_t)BATCH * HID * 2 + (size_t)BATCH * HID * 4
                       + (size_t)BATCH * H3 * 4;   // 147456 B
  k_gru_seq<<<1, 1024, gru_lds, stream>>>(gi0f, w_whh0, enc_bhh0,
      y0_16, nullptr, 2 * HID, hcat16, 2 * HID, 0);
  k_gru_seq<<<1, 1024, gru_lds, stream>>>(gi0b, w_whh0 + H3 * HID, enc_bhh0 + H3,
      y0_16 + HID, nullptr, 2 * HID, hcat16 + HID, 2 * HID, 1);

  // layer-1 input projections (K=512)
  gemm(y0_16, w_wih1,                 enc_bih1,      gi1f, nullptr, SB, H3, 2 * HID);
  gemm(y0_16, w_wih1 + H3 * 2 * HID,  enc_bih1 + H3, gi1b, nullptr, SB, H3, 2 * HID);

  k_gru_seq<<<1, 1024, gru_lds, stream>>>(gi1f, w_whh1, enc_bhh1,
      enc16, enc32, 2 * HID, hcat16 + 32 * 2 * HID, 2 * HID, 0);
  k_gru_seq<<<1, 1024, gru_lds, stream>>>(gi1b, w_whh1 + H3 * HID, enc_bhh1 + H3,
      enc16 + HID, enc32 + HID, 2 * HID, hcat16 + 32 * 2 * HID + HID, 2 * HID, 1);

  // decoder initial hidden: hid = concat-hiddens @ enc_out_w^T  (M=64)
  gemm(hcat16, w_eow, nullptr, hid32, hid16, 64, EMB, 2 * HID);
  // attention key precompute: att_enc = enc_out @ weight_matrix  (M=16384)
  gemm(enc16, w_wmt, nullptr, attenc, nullptr, SB, EMB, 2 * HID);

  k_fill_i32<<<1, 32, 0, stream>>>(tok, 0, BATCH);

  // ---- decoder loop (64 greedy steps) ------------------------------------
  for (int t = 0; t < TOUT; ++t) {
    gemm(h1_16, w_l3, l3_b, attdec, nullptr, BATCH, EMB, EMB);
    k_scores <<<(SB + 255) / 256, 256, 0, stream>>>(attenc, attdec, value_v, scores);
    k_softmax<<<BATCH, S_LEN, 0, stream>>>(scores, attn);
    k_xa     <<<BATCH, 2 * HID, 0, stream>>>(attn, enc32, Xa);
    k_xcat   <<<(BATCH * (EMB + 2 * HID) + 255) / 256, 256, 0, stream>>>(tok, dec_emb, Xa, xcat16);
    gemm(xcat16, w_l2, l2_b, x32, x16, BATCH, EMB, 2 * HID + EMB);

    // GRU cell 0
    gemm(x16,   w_dwih,            dec_bih,      gi_d, nullptr, BATCH, H3, EMB);
    gemm(h0_16, w_dwhh,            dec_bhh,      gh_d, nullptr, BATCH, H3, EMB);
    k_gate<<<(BATCH * HID + 255) / 256, 256, 0, stream>>>(gi_d, gh_d, h0_32, h0_16);
    // GRU cell 1
    gemm(h0_16, w_dwih + H3 * EMB, dec_bih + H3, gi_d, nullptr, BATCH, H3, EMB);
    gemm(h1_16, w_dwhh + H3 * EMB, dec_bhh + H3, gh_d, nullptr, BATCH, H3, EMB);
    k_gate<<<(BATCH * HID + 255) / 256, 256, 0, stream>>>(gi_d, gh_d, h1_32, h1_16);

    // tied output projection -> logits straight into d_out[t]
    float* logits = out + (size_t)t * BATCH * VOCAB;
    gemm(h1_16, w_demb, dec_out_b, logits, nullptr, BATCH, VOCAB, EMB);
    k_argmax<<<BATCH, 256, 0, stream>>>(logits, tok);
  }
}
